// RotarySelfAttention_40613210751421
// MI455X (gfx1250) — matrix operand order repaired and checked by reading, not compiled, not run on hardware
//
#include <hip/hip_runtime.h>
#include <hip/hip_bf16.h>

typedef __attribute__((ext_vector_type(16))) __bf16 v16bf;
typedef __attribute__((ext_vector_type(8)))  __bf16 v8bf;
typedef __attribute__((ext_vector_type(8)))  float  v8f;

#define NHEADS 16
#define HD     64
#define SEQL   2048          // L (power of two -> shift/mask, no integer division)
#define LSH    11
#define LMSK   2047

__device__ __forceinline__ v8f wmma_bf16(v16bf a, v16bf b, v8f c) {
  return __builtin_amdgcn_wmma_f32_16x16x32_bf16(false, a, false, b, (short)0, c, false, false);
}

// two contiguous 16B LDS loads -> one fragment (no per-element packing)
__device__ __forceinline__ v16bf ld16x2(const __bf16* p0, const __bf16* p1) {
  v8bf a = *reinterpret_cast<const v8bf*>(p0);
  v8bf b = *reinterpret_cast<const v8bf*>(p1);
  return __builtin_shufflevector(a, b, 0,1,2,3,4,5,6,7,8,9,10,11,12,13,14,15);
}
// A-fragment (16x32, 16-bit): lane half=0 holds K={0..7,16..23}, half=1 K={8..15,24..31}
__device__ __forceinline__ v16bf a_frag(const __bf16* row, int half) {
  return ld16x2(row + 8 * half, row + 16 + 8 * half);
}
// B-fragment (32x16, 16-bit): lane half=0 holds K={0..7,16..23}, half=1 K={8..15,24..31} (row K-contiguous)
__device__ __forceinline__ v16bf b_frag(const __bf16* row, int half) {
  return ld16x2(row + 8 * half, row + 16 + 8 * half);
}

// async DMA: 16B per lane, global -> LDS, tracked by ASYNCcnt (gfx1250)
__device__ __forceinline__ void async_ld16(void* lds, const void* gptr) {
  unsigned l = (unsigned)(uintptr_t)lds;   // generic LDS ptr: offset in addr[31:0]
  asm volatile("global_load_async_to_lds_b128 %0, %1, off"
               :: "v"(l), "v"(gptr) : "memory");
}
__device__ __forceinline__ void wait_async0() {
  asm volatile("s_wait_asynccnt 0" ::: "memory");
}

// ---------------- small prep kernels ----------------
__global__ void cvt_f32_bf16(const float* __restrict__ x, __bf16* __restrict__ y, int n) {
  int i = blockIdx.x * blockDim.x + threadIdx.x;
  if (i < n) y[i] = (__bf16)x[i];
}

// rope cache: cosT/sinT [SEQL][32]
__global__ void rope_cache(float* __restrict__ c, float* __restrict__ s) {
  int i = blockIdx.x * blockDim.x + threadIdx.x;
  if (i < SEQL * 32) {
    int d = i & 31, t = i >> 5;
    float invf = __expf(-0.28782313662425572f * (float)d);  // 10000^(-d/32)
    float ang  = (float)t * invf;
    c[i] = cosf(ang);
    s[i] = sinf(ang);
  }
}

// ---------------- GEMM: Out = A[M,K] @ W[N,K]^T + bias ----------------
// MODE 0: Q -> RoPE, *0.125 (folded attn scale), store bf16 [B,H,L,Hd]
// MODE 1: K -> RoPE, store bf16 [B,H,L,Hd]
// MODE 2: V -> store bf16 [B,H,Hd,L] (transposed: ctx-GEMM B-operand)
// MODE 3: final -> fp32 row-major [M,N]
template<int MODE>
__global__ __launch_bounds__(256)
void gemm128(const __bf16* __restrict__ A, const __bf16* __restrict__ W,
             const float* __restrict__ bias, void* __restrict__ Out,
             int M, int N, int K,
             const float* __restrict__ ropeC, const float* __restrict__ ropeS)
{
  __shared__ __bf16 As[2][128][40];   // double-buffered [m][k] (row stride 80B)
  __shared__ __bf16 Ws[2][128][40];   // double-buffered [n][k]

  const int tid  = threadIdx.x;
  const int lane = tid & 31, wave = tid >> 5;
  const int half = lane >> 4, l15 = lane & 15;
  const int wy = wave >> 1, wx = wave & 1;            // 4 x 2 wave grid
  const int bm = blockIdx.y * 128, bn = blockIdx.x * 128;

  // per-thread load slots (2 x 16B per tile per operand)
  const int lr0 = tid >> 2,            lc0 = (tid & 3) * 8;
  const int lr1 = (tid + 256) >> 2,    lc1 = ((tid + 256) & 3) * 8;

  v8f acc[2][4];
  #pragma unroll
  for (int s = 0; s < 2; ++s)
    #pragma unroll
    for (int j = 0; j < 4; ++j)
      #pragma unroll
      for (int r = 0; r < 8; ++r) acc[s][j][r] = 0.0f;

  // prologue: issue tile 0
  async_ld16(&As[0][lr0][lc0], A + (size_t)(bm + lr0) * K + lc0);
  async_ld16(&Ws[0][lr0][lc0], W + (size_t)(bn + lr0) * K + lc0);
  async_ld16(&As[0][lr1][lc1], A + (size_t)(bm + lr1) * K + lc1);
  async_ld16(&Ws[0][lr1][lc1], W + (size_t)(bn + lr1) * K + lc1);
  wait_async0();
  __syncthreads();

  const int NT = K >> 5;
  for (int kt = 0; kt < NT; ++kt) {
    const int cur = kt & 1;
    if (kt + 1 < NT) {          // overlap: DMA next tile while computing current
      const int k1 = (kt + 1) << 5;
      async_ld16(&As[cur ^ 1][lr0][lc0], A + (size_t)(bm + lr0) * K + k1 + lc0);
      async_ld16(&Ws[cur ^ 1][lr0][lc0], W + (size_t)(bn + lr0) * K + k1 + lc0);
      async_ld16(&As[cur ^ 1][lr1][lc1], A + (size_t)(bm + lr1) * K + k1 + lc1);
      async_ld16(&Ws[cur ^ 1][lr1][lc1], W + (size_t)(bn + lr1) * K + k1 + lc1);
    }

    v16bf bfr[4];
    #pragma unroll
    for (int j = 0; j < 4; ++j) bfr[j] = b_frag(&Ws[cur][wx * 64 + j * 16 + l15][0], half);
    #pragma unroll
    for (int s = 0; s < 2; ++s) {
      v16bf a = a_frag(&As[cur][wy * 32 + s * 16 + l15][0], half);
      #pragma unroll
      for (int j = 0; j < 4; ++j) acc[s][j] = wmma_bf16(a, bfr[j], acc[s][j]);
    }

    wait_async0();      // next tile resident
    __syncthreads();    // all waves done with current tile
  }

  // ---- epilogue (incremental, register-lean) ----
  const int colbase = bn + wx * 64;
  #pragma unroll
  for (int s = 0; s < 2; ++s) {
    const int rowbase = bm + wy * 32 + s * 16;
    if (MODE == 3) {
      #pragma unroll
      for (int j = 0; j < 4; ++j) {
        const int n = colbase + j * 16 + l15;
        const float bj = bias[n];
        #pragma unroll
        for (int r = 0; r < 8; ++r) {
          int gm = rowbase + r + 8 * half;
          ((float*)Out)[(size_t)gm * N + n] = acc[s][j][r] + bj;
        }
      }
    } else if (MODE == 2) {
      #pragma unroll
      for (int j = 0; j < 4; ++j) {
        const int n = colbase + j * 16 + l15;
        const int h = n >> 6, d = n & 63;
        const float bj = bias[n];
        #pragma unroll
        for (int r = 0; r < 8; ++r) {
          int gm = rowbase + r + 8 * half;
          int b = gm >> LSH, t = gm & LMSK;
          ((__bf16*)Out)[(((size_t)(b * NHEADS + h)) * HD + d) * SEQL + t] =
              (__bf16)(acc[s][j][r] + bj);
        }
      }
    } else {  // MODE 0/1: RoPE pairs (j, j+2) processed together, table-based trig
      #pragma unroll
      for (int j = 0; j < 2; ++j) {
        const int d0 = j * 16 + l15;          // 0..31
        const int n0 = colbase + d0;
        const int h  = n0 >> 6;
        const float b0 = bias[n0], b1 = bias[n0 + 32];
        #pragma unroll
        for (int r = 0; r < 8; ++r) {
          int gm = rowbase + r + 8 * half;
          int b = gm >> LSH, t = gm & LMSK;
          float c  = ropeC[t * 32 + d0];
          float sn = ropeS[t * 32 + d0];
          float lo = acc[s][j][r] + b0;
          float hi = acc[s][j + 2][r] + b1;
          float o0 = lo * c - hi * sn;
          float o1 = hi * c + lo * sn;
          if (MODE == 0) { o0 *= 0.125f; o1 *= 0.125f; }   // 1/sqrt(Hd)
          __bf16* dst = (__bf16*)Out + (((size_t)(b * NHEADS + h)) * SEQL + t) * HD;
          dst[d0]      = (__bf16)o0;
          dst[d0 + 32] = (__bf16)o1;
        }
      }
    }
  }
}

// ---------------- flash attention: 128 q-rows/block (8 waves), full L sweep ----------------
// Q,K: [B,H,L,Hd] bf16 (Q pre-scaled by 1/sqrt(Hd));  Vt: [B,H,Hd,L] bf16
__global__ __launch_bounds__(256)
void attn128(const __bf16* __restrict__ Qb, const __bf16* __restrict__ Kb,
             const __bf16* __restrict__ Vt, __bf16* __restrict__ Ctx)
{
  __shared__ __bf16 Qs[128][72];        // [tok][hd]
  __shared__ __bf16 Ks[2][64][72];      // double-buffered [tok][hd]
  __shared__ __bf16 Vs[2][64][72];      // double-buffered [hd][tok]
  __shared__ __bf16 Ps[8][16][72];      // per-wave P tile [m][n]

  const int tid  = threadIdx.x;
  const int lane = tid & 31, wave = tid >> 5;
  const int half = lane >> 4, l15 = lane & 15;
  const int bh = blockIdx.y;
  const int q0 = blockIdx.x * 128;

  const __bf16* Kg = Kb + (size_t)bh * SEQL * HD;   // [tok][hd]
  const __bf16* Vg = Vt + (size_t)bh * HD * SEQL;   // [hd][tok]

  // per-thread K/V load slots (2 x 16B per tile per operand)
  const int kr0 = tid >> 3,           kc0 = (tid & 7) * 8;
  const int kr1 = (tid + 256) >> 3,   kc1 = ((tid + 256) & 7) * 8;

  // async-load Q tile 128x64 + first K/V tile
  const __bf16* Qg = Qb + ((size_t)bh * SEQL + q0) * HD;
  #pragma unroll
  for (int j = 0; j < 4; ++j) {
    int lin = tid + j * 256;           // 0..1023
    int row = lin >> 3, col = (lin & 7) * 8;
    async_ld16(&Qs[row][col], Qg + (size_t)row * HD + col);
  }
  async_ld16(&Ks[0][kr0][kc0], Kg + (size_t)kr0 * HD + kc0);
  async_ld16(&Vs[0][kr0][kc0], Vg + (size_t)kr0 * SEQL + kc0);
  async_ld16(&Ks[0][kr1][kc1], Kg + (size_t)kr1 * HD + kc1);
  async_ld16(&Vs[0][kr1][kc1], Vg + (size_t)kr1 * SEQL + kc1);
  wait_async0();
  __syncthreads();

  v16bf qa[2];
  #pragma unroll
  for (int ks = 0; ks < 2; ++ks)
    qa[ks] = a_frag(&Qs[wave * 16 + l15][ks * 32], half);

  v8f cacc[4];
  #pragma unroll
  for (int j = 0; j < 4; ++j)
    #pragma unroll
    for (int r = 0; r < 8; ++r) cacc[j][r] = 0.0f;
  float mrow[8], lrow[8];
  #pragma unroll
  for (int r = 0; r < 8; ++r) { mrow[r] = -1e30f; lrow[r] = 0.0f; }

  for (int kt = 0; kt < SEQL; kt += 64) {
    const int cur = (kt >> 6) & 1;
    if (kt + 64 < SEQL) {        // overlap: DMA next K/V tile while computing current
      const int nb = cur ^ 1, kn = kt + 64;
      async_ld16(&Ks[nb][kr0][kc0], Kg + (size_t)(kn + kr0) * HD + kc0);
      async_ld16(&Vs[nb][kr0][kc0], Vg + (size_t)kr0 * SEQL + kn + kc0);
      async_ld16(&Ks[nb][kr1][kc1], Kg + (size_t)(kn + kr1) * HD + kc1);
      async_ld16(&Vs[nb][kr1][kc1], Vg + (size_t)kr1 * SEQL + kn + kc1);
    }

    // S = Q @ K^T : B[k=hd][n=tok] = Ks[tok][hd] -> K-contiguous rows
    v8f sacc[4];
    #pragma unroll
    for (int j = 0; j < 4; ++j) {
      #pragma unroll
      for (int r = 0; r < 8; ++r) sacc[j][r] = 0.0f;
      #pragma unroll
      for (int ks = 0; ks < 2; ++ks) {
        v16bf kbf = b_frag(&Ks[cur][j * 16 + l15][ks * 32], half);
        sacc[j] = wmma_bf16(qa[ks], kbf, sacc[j]);
      }
    }

    // online softmax (row m = r + 8*half lives in one 16-lane half)
    #pragma unroll
    for (int r = 0; r < 8; ++r) {
      float mx = fmaxf(fmaxf(sacc[0][r], sacc[1][r]), fmaxf(sacc[2][r], sacc[3][r]));
      #pragma unroll
      for (int off = 1; off < 16; off <<= 1) mx = fmaxf(mx, __shfl_xor(mx, off, 32));
      float mn = fmaxf(mrow[r], mx);
      float corr = __expf(mrow[r] - mn);
      float p0 = __expf(sacc[0][r] - mn);
      float p1 = __expf(sacc[1][r] - mn);
      float p2 = __expf(sacc[2][r] - mn);
      float p3 = __expf(sacc[3][r] - mn);
      float rs = p0 + p1 + p2 + p3;
      #pragma unroll
      for (int off = 1; off < 16; off <<= 1) rs += __shfl_xor(rs, off, 32);
      lrow[r] = lrow[r] * corr + rs;
      mrow[r] = mn;
      #pragma unroll
      for (int j = 0; j < 4; ++j) cacc[j][r] *= corr;
      int m = r + 8 * half;
      Ps[wave][m][ 0 + l15] = (__bf16)p0;
      Ps[wave][m][16 + l15] = (__bf16)p1;
      Ps[wave][m][32 + l15] = (__bf16)p2;
      Ps[wave][m][48 + l15] = (__bf16)p3;
    }

    // ctx += P @ V : B[k=tok][n=hd] = Vs[hd][tok] -> K-contiguous rows
    #pragma unroll
    for (int ks = 0; ks < 2; ++ks) {
      v16bf pa = a_frag(&Ps[wave][l15][ks * 32], half);
      #pragma unroll
      for (int j = 0; j < 4; ++j) {
        v16bf vf = b_frag(&Vs[cur][j * 16 + l15][ks * 32], half);
        cacc[j] = wmma_bf16(pa, vf, cacc[j]);
      }
    }

    wait_async0();      // next K/V tile resident
    __syncthreads();    // all waves done with current tile
  }

  const int b = bh >> 4, h = bh & (NHEADS - 1);
  #pragma unroll
  for (int r = 0; r < 8; ++r) {
    int m = r + 8 * half;
    int t = q0 + wave * 16 + m;
    float inv = 1.0f / lrow[r];
    #pragma unroll
    for (int j = 0; j < 4; ++j) {
      float v = cacc[j][r] * inv;
      Ctx[((size_t)b * SEQL + t) * (NHEADS * HD) + h * HD + j * 16 + l15] = (__bf16)v;
    }
  }
}

extern "C" void kernel_launch(void* const* d_in, const int* in_sizes, int n_in,
                              void* d_out, int out_size, void* d_ws, size_t ws_size,
                              hipStream_t stream) {
  const float* x  = (const float*)d_in[0];
  const float* Wq = (const float*)d_in[1]; const float* bq = (const float*)d_in[2];
  const float* Wk = (const float*)d_in[3]; const float* bk = (const float*)d_in[4];
  const float* Wv = (const float*)d_in[5]; const float* bv = (const float*)d_in[6];
  const float* Wo = (const float*)d_in[7]; const float* bo = (const float*)d_in[8];

  const int B = 2, L = SEQL, D = 1024;
  const int M = B * L;

  char* ws = (char*)d_ws;
  __bf16* xb   = (__bf16*)ws; ws += (size_t)M * D * sizeof(__bf16);
  __bf16* Wqb  = (__bf16*)ws; ws += (size_t)D * D * sizeof(__bf16);
  __bf16* Wkb  = (__bf16*)ws; ws += (size_t)D * D * sizeof(__bf16);
  __bf16* Wvb  = (__bf16*)ws; ws += (size_t)D * D * sizeof(__bf16);
  __bf16* Wob  = (__bf16*)ws; ws += (size_t)D * D * sizeof(__bf16);
  __bf16* Qb   = (__bf16*)ws; ws += (size_t)M * D * sizeof(__bf16);
  __bf16* Kb   = (__bf16*)ws; ws += (size_t)M * D * sizeof(__bf16);
  __bf16* Vtb  = (__bf16*)ws; ws += (size_t)M * D * sizeof(__bf16);
  __bf16* Ctxb = (__bf16*)ws; ws += (size_t)M * D * sizeof(__bf16);
  float*  ropeC = (float*)ws; ws += (size_t)L * 32 * sizeof(float);
  float*  ropeS = (float*)ws; ws += (size_t)L * 32 * sizeof(float);

  cvt_f32_bf16<<<(M * D + 255) / 256, 256, 0, stream>>>(x,  xb,  M * D);
  cvt_f32_bf16<<<(D * D + 255) / 256, 256, 0, stream>>>(Wq, Wqb, D * D);
  cvt_f32_bf16<<<(D * D + 255) / 256, 256, 0, stream>>>(Wk, Wkb, D * D);
  cvt_f32_bf16<<<(D * D + 255) / 256, 256, 0, stream>>>(Wv, Wvb, D * D);
  cvt_f32_bf16<<<(D * D + 255) / 256, 256, 0, stream>>>(Wo, Wob, D * D);
  rope_cache<<<(L * 32 + 255) / 256, 256, 0, stream>>>(ropeC, ropeS);

  dim3 gg(D / 128, M / 128), blk(256);
  gemm128<0><<<gg, blk, 0, stream>>>(xb, Wqb, bq, (void*)Qb,  M, D, D, ropeC, ropeS);
  gemm128<1><<<gg, blk, 0, stream>>>(xb, Wkb, bk, (void*)Kb,  M, D, D, ropeC, ropeS);
  gemm128<2><<<gg, blk, 0, stream>>>(xb, Wvb, bv, (void*)Vtb, M, D, D, nullptr, nullptr);

  attn128<<<dim3(L / 128, B * NHEADS), dim3(256), 0, stream>>>(Qb, Kb, Vtb, Ctxb);

  gemm128<3><<<gg, blk, 0, stream>>>(Ctxb, Wob, bo, d_out, M, D, D, nullptr, nullptr);
}
